// CrossModalityMultiHeadAttention_50757923504474
// MI455X (gfx1250) — matrix operand order repaired and checked by reading, not compiled, not run on hardware
//
#include <hip/hip_runtime.h>
#include <hip/hip_bf16.h>

// ---------------------------------------------------------------------------
// CDNA5 (gfx1250) wave32 WMMA implementation of cross-modality MHA.
// All matmuls: v_wmma_f32_16x16x32_bf16 (f32 accumulate).
// - every WMMA operand is a pair of 16B contiguous loads per lane
// - independent GEMMs / both attention directions batched via blockIdx.z
// - __launch_bounds__ sized so the scheduler never spills accumulators
// ---------------------------------------------------------------------------

typedef __attribute__((ext_vector_type(16))) __bf16 v16bf;
typedef __attribute__((ext_vector_type(8)))  __bf16 v8bf;
typedef __attribute__((ext_vector_type(8)))  float  v8f;

#define WMMA_BF16(a, b, c) \
  __builtin_amdgcn_wmma_f32_16x16x32_bf16(false, (a), false, (b), (short)0, (c), false, false)

#define HID 512
#define NHEAD 8
#define DHEAD 64
#define NEGV  -90000000000.0f

// ---- fragment loaders -----------------------------------------------------
// A fragment (16x32, 16-bit): lane m = lane&15 is the row; lanes 0-15 hold
// K = 0..15, lanes 16-31 hold K = 16..31 (contiguous, matches B).
__device__ inline v16bf load_a16(const __bf16* base, int hi) {
  const int kb = hi * 16;
  v8bf lo = *reinterpret_cast<const v8bf*>(base + kb);
  v8bf hh = *reinterpret_cast<const v8bf*>(base + 8 + kb);
  v16bf r;
#pragma unroll
  for (int i = 0; i < 8; ++i) { r[i] = lo[i]; r[8 + i] = hh[i]; }
  return r;
}

// B fragment (32x16, 16-bit): lane n = lane&15 is the column; lanes 0-15 hold
// K=0..15, lanes 16-31 hold K=16..31, contiguous in memory at p.
__device__ inline v16bf load_b16_contig(const __bf16* p) {
  v8bf lo = *reinterpret_cast<const v8bf*>(p);
  v8bf hh = *reinterpret_cast<const v8bf*>(p + 8);
  v16bf r;
#pragma unroll
  for (int i = 0; i < 8; ++i) { r[i] = lo[i]; r[8 + i] = hh[i]; }
  return r;
}

// ---- f32 -> bf16 activation conversion with column zero-padding -----------
__global__ void cvt_pad_kernel(const float* __restrict__ src, __bf16* __restrict__ dst,
                               int rows, int cols, int padCols) {
  int idx = blockIdx.x * blockDim.x + threadIdx.x;
  if (idx >= rows * padCols) return;
  int r = idx / padCols, c = idx - r * padCols;
  dst[idx] = (c < cols) ? (__bf16)src[r * cols + c] : (__bf16)0.0f;
}

// ---- f32 W[Kw x 512] -> bf16 W^T[512 x Kpad] (zero pad K) ------------------
__global__ void cvt_wT_kernel(const float* __restrict__ W, __bf16* __restrict__ Wt,
                              int Kw, int Kpad) {
  int idx = blockIdx.x * blockDim.x + threadIdx.x;
  if (idx >= HID * Kpad) return;
  int c = idx / Kpad, k = idx - c * Kpad;
  Wt[idx] = (k < Kw) ? (__bf16)W[(size_t)k * HID + c] : (__bf16)0.0f;
}

// batched transpose for the 8 square 512x512 weights
struct CvtW8 { const float* src[8]; __bf16* dst[8]; };
__global__ void cvt_wT8_kernel(CvtW8 P) {
  const int i = blockIdx.y;
  int idx = blockIdx.x * blockDim.x + threadIdx.x;
  if (idx >= HID * HID) return;
  int c = idx >> 9, k = idx & (HID - 1);
  P.dst[i][idx] = (__bf16)P.src[i][(size_t)k * HID + c];
}

// ---- batched GEMM ----------------------------------------------------------
// out[N x 512] = A_bf16[N x K] * Wt^T + bias.  blockIdx.z selects the job.
// block = 256 threads = 8 waves; wave tile = 32 rows x 64 cols (8 accums).
// Two A fragments share four B fragments -> 8 WMMA per 6 fragment-loads.
struct GemmJob {
  const __bf16* A;
  const __bf16* Wt;
  const float*  bias;
  __bf16* outBf;
  float*  outF;
  __bf16* outBfT;
  int N, ldt;
};
struct GemmBatch { GemmJob j[6]; };

__global__ __launch_bounds__(256, 2)
void gemm_bf16_kernel(GemmBatch B_, int lda, int K, int ldk) {
  const GemmJob job = B_.j[blockIdx.z];
  const int lane = threadIdx.x & 31;
  const int wid  = threadIdx.x >> 5;
  const int row0 = blockIdx.x * 128 + (wid & 3) * 32;
  const int col0 = blockIdx.y * 128 + (wid >> 2) * 64;
  if (row0 >= job.N) return;
  const int n = lane & 15, hi = lane >> 4;
  const int kb16 = hi ? 16 : 0;

  const __bf16* Arow0 = job.A + (size_t)(row0 + n) * lda;
  const __bf16* Arow1 = job.A + (size_t)(row0 + 16 + n) * lda;
  const __bf16* Wcol  = job.Wt + (size_t)(col0 + n) * ldk + kb16;

  v8f z = {};
  v8f acc[2][4] = {{z, z, z, z}, {z, z, z, z}};

  for (int k0 = 0; k0 < K; k0 += 32) {
    if (k0 + 64 < K) {
      __builtin_prefetch(Arow0 + k0 + 64, 0, 1);
      __builtin_prefetch(Arow1 + k0 + 64, 0, 1);
    }
    v16bf a0 = load_a16(Arow0 + k0, hi);
    v16bf a1 = load_a16(Arow1 + k0, hi);
#pragma unroll
    for (int t = 0; t < 4; ++t) {
      v16bf b = load_b16_contig(Wcol + (size_t)t * 16 * ldk + k0);
      acc[0][t] = WMMA_BF16(a0, b, acc[0][t]);
      acc[1][t] = WMMA_BF16(a1, b, acc[1][t]);
    }
  }

#pragma unroll
  for (int rt = 0; rt < 2; ++rt)
#pragma unroll
    for (int t = 0; t < 4; ++t) {
      const int c = col0 + t * 16 + n;
      const float bv = job.bias[c];
#pragma unroll
      for (int j = 0; j < 8; ++j) {
        const int r = row0 + rt * 16 + j + 8 * hi;
        const float v = acc[rt][t][j] + bv;
        if (job.outBf)  job.outBf[(size_t)r * HID + c]      = (__bf16)v;
        if (job.outF)   job.outF[(size_t)r * HID + c]       = v;
        if (job.outBfT) job.outBfT[(size_t)c * job.ldt + r] = (__bf16)v;
      }
    }
}

// ---- streaming-softmax cross attention (both directions in one launch) ----
// wave = one (side, head, 32-query-row tile); 32 keys per iteration.
// K fragments and V fragments are shared by both 16-row sub-tiles.
// ctx[r][h*64+d] = Vq[r][h*64+d] + softmax(mask-hacked S) @ V
struct AttnJob {
  const __bf16* Q;
  const __bf16* Km;
  const __bf16* VmT;  // [512][Nk]
  const __bf16* Vq;
  const int* qbatch;
  const int* kbatch;
  __bf16* ctx;
  int Nq, Nk;
};
struct AttnPair { AttnJob j[2]; };

__global__ __launch_bounds__(128, 2)
void attn_kernel(AttnPair P) {
  __shared__ __bf16 plds[4][2][16 * 32];  // per-wave P tiles

  const AttnJob job = P.j[blockIdx.z];
  const int lane = threadIdx.x & 31;
  const int wid  = threadIdx.x >> 5;
  const int h    = blockIdx.y;
  const int q0   = (blockIdx.x * 4 + wid) * 32;
  if (q0 >= job.Nq) return;  // wave-uniform
  const int n = lane & 15, hi = lane >> 4;
  const int hb = h * DHEAD;
  const int kb16 = hi ? 16 : 0;
  const int Nk = job.Nk;

  v16bf qf[2][2];
  int   qb[2][8];
  float rmax[2][8], rsum[2][8];
  v8f z = {};
  v8f O[2][4];
#pragma unroll
  for (int rt = 0; rt < 2; ++rt) {
    const __bf16* qrow = job.Q + (size_t)(q0 + rt * 16 + n) * HID + hb;
    qf[rt][0] = load_a16(qrow, hi);
    qf[rt][1] = load_a16(qrow + 32, hi);
#pragma unroll
    for (int e = 0; e < 8; ++e) {
      qb[rt][e]   = job.qbatch[q0 + rt * 16 + e + 8 * hi];
      rmax[rt][e] = -INFINITY;
      rsum[rt][e] = 0.0f;
    }
#pragma unroll
    for (int t = 0; t < 4; ++t) O[rt][t] = z;
  }

  for (int kt = 0; kt < Nk; kt += 32) {
    // ---- K^T fragments for 2 half-tiles of 16 keys, 2 k-steps each
    const __bf16* kp0 = job.Km + (size_t)(kt + n) * HID + hb + kb16;
    const __bf16* kp1 = job.Km + (size_t)(kt + 16 + n) * HID + hb + kb16;
    const v16bf b00 = load_b16_contig(kp0);
    const v16bf b01 = load_b16_contig(kp0 + 32);
    const v16bf b10 = load_b16_contig(kp1);
    const v16bf b11 = load_b16_contig(kp1 + 32);
    const int kb0 = job.kbatch[kt + n];
    const int kb1 = job.kbatch[kt + 16 + n];

#pragma unroll
    for (int rt = 0; rt < 2; ++rt) {
      v8f s0 = z, s1 = z;
      s0 = WMMA_BF16(qf[rt][0], b00, s0);
      s0 = WMMA_BF16(qf[rt][1], b01, s0);
      s1 = WMMA_BF16(qf[rt][0], b10, s1);
      s1 = WMMA_BF16(qf[rt][1], b11, s1);

      __bf16* pl = plds[wid][rt];
#pragma unroll
      for (int e = 0; e < 8; ++e) {
        // faithful reference hack: s = dot*mask/8; (s==0) -> NEG
        float sv0 = s0[e] * 0.125f; sv0 = (qb[rt][e] == kb0) ? sv0 : 0.0f; if (sv0 == 0.0f) sv0 = NEGV;
        float sv1 = s1[e] * 0.125f; sv1 = (qb[rt][e] == kb1) ? sv1 : 0.0f; if (sv1 == 0.0f) sv1 = NEGV;
        // row reductions: xor 1/2/4/8 butterflies never mix the two row halves
        float tm = fmaxf(sv0, sv1);
        tm = fmaxf(tm, __shfl_xor(tm, 1, 32));
        tm = fmaxf(tm, __shfl_xor(tm, 2, 32));
        tm = fmaxf(tm, __shfl_xor(tm, 4, 32));
        tm = fmaxf(tm, __shfl_xor(tm, 8, 32));
        const float nm = fmaxf(rmax[rt][e], tm);
        const float sc = __expf(rmax[rt][e] - nm);
        const float p0 = __expf(sv0 - nm);
        const float p1 = __expf(sv1 - nm);
        float ts = p0 + p1;
        ts += __shfl_xor(ts, 1, 32);
        ts += __shfl_xor(ts, 2, 32);
        ts += __shfl_xor(ts, 4, 32);
        ts += __shfl_xor(ts, 8, 32);
        rsum[rt][e] = rsum[rt][e] * sc + ts;
        rmax[rt][e] = nm;
#pragma unroll
        for (int t = 0; t < 4; ++t) O[rt][t][e] *= sc;   // online rescale
        const int r = e + 8 * hi;
        pl[r * 32 + n]      = (__bf16)p0;
        pl[r * 32 + 16 + n] = (__bf16)p1;
      }
    }
    asm volatile("s_wait_dscnt 0" ::: "memory");

    // ---- reshape P (C-layout) -> A-fragment layout via LDS
    v16bf pa[2];
#pragma unroll
    for (int rt = 0; rt < 2; ++rt) {
      const __bf16* pl = plds[wid][rt];
      const int kb = hi * 16;
#pragma unroll
      for (int i = 0; i < 8; ++i) {
        pa[rt][i]     = pl[n * 32 + kb + i];
        pa[rt][8 + i] = pl[n * 32 + 8 + kb + i];
      }
    }

    // ---- O += P * V  (V fragments shared by both row tiles)
#pragma unroll
    for (int t = 0; t < 4; ++t) {
      const __bf16* vp = job.VmT + (size_t)(hb + t * 16 + n) * Nk + kt + kb16;
      const v16bf vb = load_b16_contig(vp);
      O[0][t] = WMMA_BF16(pa[0], vb, O[0][t]);
      O[1][t] = WMMA_BF16(pa[1], vb, O[1][t]);
    }
  }

  // epilogue: normalize + residual add of raw query-side values
#pragma unroll
  for (int rt = 0; rt < 2; ++rt)
#pragma unroll
    for (int t = 0; t < 4; ++t)
#pragma unroll
      for (int e = 0; e < 8; ++e) {
        const int r = q0 + rt * 16 + e + 8 * hi;
        const int c = hb + t * 16 + n;
        const float v = O[rt][t][e] / rsum[rt][e] + (float)job.Vq[(size_t)r * HID + c];
        job.ctx[(size_t)r * HID + c] = (__bf16)v;
      }
}

// ---------------------------------------------------------------------------
extern "C" void kernel_launch(void* const* d_in, const int* in_sizes, int n_in,
                              void* d_out, int out_size, void* d_ws, size_t ws_size,
                              hipStream_t stream) {
  const float* h_head  = (const float*)d_in[0];
  const float* h_tail  = (const float*)d_in[1];
  const int*   b_head  = (const int*)d_in[2];
  const int*   b_tail  = (const int*)d_in[3];
  const float* map_w   = (const float*)d_in[4];
  const float* map_b   = (const float*)d_in[5];
  const float* q1_w = (const float*)d_in[6],  *q1_b = (const float*)d_in[7];
  const float* k1_w = (const float*)d_in[8],  *k1_b = (const float*)d_in[9];
  const float* v1_w = (const float*)d_in[10], *v1_b = (const float*)d_in[11];
  const float* q2_w = (const float*)d_in[12], *q2_b = (const float*)d_in[13];
  const float* k2_w = (const float*)d_in[14], *k2_b = (const float*)d_in[15];
  const float* v2_w = (const float*)d_in[16], *v2_b = (const float*)d_in[17];
  const float* d1_w = (const float*)d_in[18], *d1_b = (const float*)d_in[19];
  const float* d2_w = (const float*)d_in[20], *d2_b = (const float*)d_in[21];
  float* out = (float*)d_out;

  const int D1 = 300, D1P = 320;
  const int N1 = in_sizes[0] / D1;   // 4096
  const int N2 = in_sizes[1] / HID;  // 4096

  // workspace carve-out
  char* w = (char*)d_ws;
  auto carve = [&](size_t bytes) {
    char* p = w;
    w += (bytes + 255) & ~(size_t)255;
    return p;
  };
  __bf16* A_head = (__bf16*)carve((size_t)N1 * D1P * 2);
  __bf16* A_tail = (__bf16*)carve((size_t)N2 * HID * 2);
  __bf16* hh  = (__bf16*)carve((size_t)N1 * HID * 2);
  __bf16* qh  = (__bf16*)carve((size_t)N1 * HID * 2);
  __bf16* kh  = (__bf16*)carve((size_t)N1 * HID * 2);
  __bf16* vh  = (__bf16*)carve((size_t)N1 * HID * 2);
  __bf16* qt  = (__bf16*)carve((size_t)N2 * HID * 2);
  __bf16* kt  = (__bf16*)carve((size_t)N2 * HID * 2);
  __bf16* vt  = (__bf16*)carve((size_t)N2 * HID * 2);
  __bf16* vhT = (__bf16*)carve((size_t)HID * N1 * 2);
  __bf16* vtT = (__bf16*)carve((size_t)HID * N2 * 2);
  __bf16* cx0 = (__bf16*)carve((size_t)N1 * HID * 2);
  __bf16* cx1 = (__bf16*)carve((size_t)N2 * HID * 2);
  __bf16* wtm = (__bf16*)carve((size_t)HID * D1P * 2);  // map_w^T
  __bf16* wtx[8];
  for (int i = 0; i < 8; ++i) wtx[i] = (__bf16*)carve((size_t)HID * HID * 2);
  (void)ws_size; (void)n_in; (void)out_size;

  // 1) convert activations to bf16 (head K padded 300 -> 320 with zeros)
  cvt_pad_kernel<<<(N1 * D1P + 255) / 256, 256, 0, stream>>>(h_head, A_head, N1, D1, D1P);
  cvt_pad_kernel<<<(N2 * HID + 255) / 256, 256, 0, stream>>>(h_tail, A_tail, N2, HID, HID);

  // 2) transpose + convert all weights to bf16 (once; weights are tiny)
  cvt_wT_kernel<<<(HID * D1P + 255) / 256, 256, 0, stream>>>(map_w, wtm, D1, D1P);
  {
    CvtW8 cw;
    const float* ws_[8] = {q1_w, k1_w, v1_w, q2_w, k2_w, v2_w, d1_w, d2_w};
    for (int i = 0; i < 8; ++i) { cw.src[i] = ws_[i]; cw.dst[i] = wtx[i]; }
    cvt_wT8_kernel<<<dim3((HID * HID + 255) / 256, 8), 256, 0, stream>>>(cw);
  }

  const dim3 gblk(256);
  // 3) hh = h_head @ map_w + map_b
  {
    GemmBatch b{};
    b.j[0] = {A_head, wtm, map_b, hh, nullptr, nullptr, N1, 0};
    gemm_bf16_kernel<<<dim3(N1 / 128, HID / 128, 1), gblk, 0, stream>>>(b, D1P, D1P, D1P);
  }
  // 4) all six QKV projections in one launch (V also written transposed)
  {
    GemmBatch b{};
    b.j[0] = {hh,     wtx[0], q1_b, qh, nullptr, nullptr, N1, 0};
    b.j[1] = {hh,     wtx[1], k1_b, kh, nullptr, nullptr, N1, 0};
    b.j[2] = {hh,     wtx[2], v1_b, vh, nullptr, vhT,     N1, N1};
    b.j[3] = {A_tail, wtx[3], q2_b, qt, nullptr, nullptr, N2, 0};
    b.j[4] = {A_tail, wtx[4], k2_b, kt, nullptr, nullptr, N2, 0};
    b.j[5] = {A_tail, wtx[5], v2_b, vt, nullptr, vtT,     N2, N2};
    const int nmax = (N1 > N2) ? N1 : N2;
    gemm_bf16_kernel<<<dim3(nmax / 128, HID / 128, 6), gblk, 0, stream>>>(b, HID, HID, HID);
  }
  // 5) both cross-attention directions in one launch
  {
    AttnPair p{};
    p.j[0] = {qh, kt, vtT, vh, b_head, b_tail, cx0, N1, N2};
    p.j[1] = {qt, kh, vhT, vt, b_tail, b_head, cx1, N2, N1};
    const int nmax = (N1 > N2) ? N1 : N2;
    attn_kernel<<<dim3(nmax / 128, NHEAD, 2), dim3(128), 0, stream>>>(p);
  }
  // 6) both output projections in one launch -> f32 d_out (head then tail)
  {
    GemmBatch b{};
    b.j[0] = {cx0, wtx[6], d1_b, nullptr, out,                      nullptr, N1, 0};
    b.j[1] = {cx1, wtx[7], d2_b, nullptr, out + (size_t)N1 * HID,   nullptr, N2, 0};
    const int nmax = (N1 > N2) ? N1 : N2;
    gemm_bf16_kernel<<<dim3(nmax / 128, HID / 128, 2), gblk, 0, stream>>>(b, HID, HID, HID);
  }
}
